// SentenceT5_MLP_Agg_Hyb_Prior_60438779789379
// MI455X (gfx1250) — matrix operand order repaired and checked by reading, not compiled, not run on hardware
//
#include <hip/hip_runtime.h>
#include <hip/hip_bf16.h>
#include <stdint.h>

// ---- CDNA5 WMMA vector types ----
typedef __attribute__((ext_vector_type(16))) __bf16 v16bf;
typedef __attribute__((ext_vector_type(8)))  float  v8f;
typedef __attribute__((ext_vector_type(4)))  unsigned v4u;
typedef __attribute__((ext_vector_type(8)))  unsigned v8u;

union Frag { uint4 q[2]; unsigned int u[8]; unsigned short us[16]; __bf16 h[16]; v16bf v; };

__device__ __forceinline__ unsigned pack_bf2(float a, float b) {
  union { __bf16 h[2]; unsigned u; } p;
  p.h[0] = (__bf16)a; p.h[1] = (__bf16)b;   // native gfx1250 f32->bf16 converts
  return p.u;
}
__device__ __forceinline__ float gelu_exact(float x) {
  return 0.5f * x * (1.0f + erff(x * 0.70710678118654752f));
}

#define EMBD 768

// =====================================================================
// Kernel 1: convert+transpose W1 -> W1T bf16 [768 x 1536], W2 -> W2T bf16 [384 x 768]
// =====================================================================
__global__ __launch_bounds__(256) void convert_w_kernel(
    const float* __restrict__ W1, const float* __restrict__ W2,
    unsigned short* __restrict__ w1t, unsigned short* __restrict__ w2t) {
  const int n1 = 768 * 1536, n2 = 384 * 768;
  int idx = blockIdx.x * 256 + threadIdx.x;
  if (idx < n1) {
    int o = idx / 1536, i = idx % 1536;                    // w1t[o][i] = W1[i][o]
    union { __bf16 h; unsigned short u; } c; c.h = (__bf16)W1[(size_t)i * 768 + o];
    w1t[idx] = c.u;
  } else if (idx < n1 + n2) {
    int j = idx - n1;
    int n = j / 768, k = j % 768;                          // w2t[n][k] = W2[k][n]
    union { __bf16 h; unsigned short u; } c; c.h = (__bf16)W2[(size_t)k * 384 + n];
    w2t[j] = c.u;
  }
}

// =====================================================================
// Kernel 2: h[b, 0..63, :]  = doc[b] @ W1[:768]
//           h[b, 64..95, :] = sum[b] @ W1[768:]
// Block = (b, rowtile). A row-tile [16x768] converted to bf16 in LDS once,
// then 8 waves x (3 groups of 2 coltiles) with 2 accumulators per wave so
// each A fragment feeds 2 v_wmma_f32_16x16x32_bf16.
// =====================================================================
#define A2_STRIDE 776
__global__ __launch_bounds__(256) void h_gemm_kernel(
    const float* __restrict__ doc, const float* __restrict__ sum,
    const unsigned short* __restrict__ w1t, float* __restrict__ h) {
  __shared__ unsigned short As[16 * A2_STRIDE];            // 24832 B
  const int tid = threadIdx.x;
  const int b = blockIdx.x, rt = blockIdx.y;               // 32 x 6
  const float* abase;
  int koff, rowbase;
  if (rt < 4) { abase = doc + ((size_t)b * 64 + rt * 16) * EMBD; koff = 0;   rowbase = rt * 16; }
  else        { abase = sum + ((size_t)b * 32 + (rt - 4) * 16) * EMBD; koff = 768; rowbase = 64 + (rt - 4) * 16; }

  // ---- stage: 16x768 f32 -> bf16 LDS ----
  for (int it = 0; it < 12; ++it) {                        // 12288 elems / (256*4)
    int e = (it * 256 + tid) * 4;
    int r = e / EMBD, k = e % EMBD;
    float4 x = *(const float4*)(abase + (size_t)r * EMBD + k);
    uint2 p;
    p.x = pack_bf2(x.x, x.y);
    p.y = pack_bf2(x.z, x.w);
    *(uint2*)(As + r * A2_STRIDE + k) = p;
  }
  __syncthreads();

  // ---- GEMM: 48 coltiles over 8 waves ----
  const int wave = tid >> 5, lane = tid & 31;
  const int half = lane >> 4, m = lane & 15;
  const unsigned short* arow = As + m * A2_STRIDE;
  for (int g = 0; g < 3; ++g) {
    const int ct0 = wave + 8 * (2 * g);
    const int ct1 = wave + 8 * (2 * g + 1);
    const uint4* brow0 = (const uint4*)(w1t + (size_t)(ct0 * 16 + m) * 1536 + koff);
    const uint4* brow1 = (const uint4*)(w1t + (size_t)(ct1 * 16 + m) * 1536 + koff);
    v8f acc0 = {0.f,0.f,0.f,0.f,0.f,0.f,0.f,0.f};
    v8f acc1 = {0.f,0.f,0.f,0.f,0.f,0.f,0.f,0.f};
#pragma unroll 4
    for (int k = 0; k < 24; ++k) {
      const int kk = k * 32;
      Frag au, bu0, bu1;
      au.q[0] = *(const uint4*)(arow + kk + half * 16);        // A: 16 contiguous K per half
      au.q[1] = *(const uint4*)(arow + kk + half * 16 + 8);
      const int bb = kk + half * 16;                            // B: 16 contiguous K per half
      bu0.q[0] = brow0[bb >> 3];  bu0.q[1] = brow0[(bb >> 3) + 1];
      bu1.q[0] = brow1[bb >> 3];  bu1.q[1] = brow1[(bb >> 3) + 1];
      acc0 = __builtin_amdgcn_wmma_f32_16x16x32_bf16(false, au.v, false, bu0.v, (short)0, acc0, false, false);
      acc1 = __builtin_amdgcn_wmma_f32_16x16x32_bf16(false, au.v, false, bu1.v, (short)0, acc1, false, false);
    }
    float* hb0 = h + ((size_t)b * 96 + rowbase + 8 * half) * EMBD + ct0 * 16 + m;
    float* hb1 = h + ((size_t)b * 96 + rowbase + 8 * half) * EMBD + ct1 * 16 + m;
#pragma unroll
    for (int r = 0; r < 8; ++r) { hb0[r * EMBD] = acc0[r]; hb1[r * EMBD] = acc1[r]; }
  }
}

// =====================================================================
// Kernel 3: fused per-(b,s):  A = gelu(h_doc + h_seg_row + b1) as bf16 in LDS,
//   h2 = gelu(A @ W2 + b2), score = sigmoid(h2 @ W3 + b3), masked max -> pred,
//   cosine-prior diff.
// Staging: h_seg row via TDM (tensor_load_to_lds + s_wait_tensorcnt),
//          sum row via async global->LDS (+ s_wait_asynccnt).
// GEMM: each wave owns 3 coltiles, keeps 4 rowtile accumulators so each B
// fragment (global, L2-resident W2T) is reused by 4 WMMAs.
// =====================================================================
#define HS_OFF 0                    // 768 f32: h_seg row (+ b1)
#define SS_OFF 3072                 // 768 f32: raw sum row (for cosine)
#define SC_OFF 6144                 // 64 f32 : score accumulators
#define WM_OFF 6400                 // 8 f32  : per-wave cosine maxima
#define PV_OFF 6432                 // 1 f32  : pred value
#define A_OFF  6464                 // 64 x 776 bf16 tile
#define A_STRIDE 776
#define FUSED_LDS (A_OFF + 64 * A_STRIDE * 2)   // 105792 bytes (CDNA5 WGP: 320KB)

__global__ __launch_bounds__(256) void fused_kernel(
    const float* __restrict__ h, const float* __restrict__ doc, const float* __restrict__ sum,
    const unsigned short* __restrict__ w2t,
    const int* __restrict__ ndocp, const int* __restrict__ nsegp,
    const float* __restrict__ b1, const float* __restrict__ b2,
    const float* __restrict__ W3, const float* __restrict__ b3,
    float* __restrict__ out, float* __restrict__ diffws) {
  const int tid = threadIdx.x;
  const int b = blockIdx.x, s = blockIdx.y;
  extern __shared__ char smem[];
  float* hsf   = (float*)(smem + HS_OFF);
  float* ssf   = (float*)(smem + SS_OFF);
  float* score = (float*)(smem + SC_OFF);
  float* wmaxa = (float*)(smem + WM_OFF);
  float* pv    = (float*)(smem + PV_OFF);
  unsigned short* A = (unsigned short*)(smem + A_OFF);

  const float* hsrow = h   + ((size_t)b * 96 + 64 + s) * EMBD;
  const float* ssrow = sum + ((size_t)b * 32 + s) * EMBD;

  // ---- async stage: raw sum row -> LDS ----
  if (tid < 192) {                            // 192 lanes x 16B = 3072B
    unsigned l1 = (unsigned)(size_t)(smem + SS_OFF) + tid * 16;
    const float* g1p = ssrow + tid * 4;
    asm volatile("global_load_async_to_lds_b128 %0, %1, off" :: "v"(l1), "v"(g1p) : "memory");
  }

  // ---- TDM stage: h_seg row -> LDS via Tensor Data Mover (wave 0 issues) ----
  if (tid < 32) {
    unsigned lds_hs = (unsigned)(size_t)(smem + HS_OFF);
    unsigned glo = (unsigned)(size_t)hsrow;
    unsigned ghi = (unsigned)(((size_t)hsrow) >> 32);
    v4u g0;
    g0[0] = 1u;                                   // count=1 (valid user D#)
    g0[1] = lds_hs;                               // lds_addr (bytes)
    g0[2] = glo;                                  // global_addr[31:0]
    g0[3] = (ghi & 0x01FFFFFFu) | (2u << 30);     // global_addr[56:32] | type=2
    v8u g1;
    g1[0] = (2u << 16);                           // wg_mask=0 | data_size=4B
    g1[1] = (768u << 16);                         // tensor_dim0 = 768 (lo16)
    g1[2] = (1u << 16);                           // tensor_dim0 hi=0 | tensor_dim1 = 1
    g1[3] = (768u << 16);                         // tensor_dim1 hi=0 | tile_dim0 = 768
    g1[4] = 1u;                                   // tile_dim1 = 1 | tile_dim2 = 0
    g1[5] = 768u;                                 // tensor_dim0_stride[31:0]
    g1[6] = 0u;                                   // stride hi | dim1_stride lo
    g1[7] = 0u;
    asm volatile("tensor_load_to_lds %0, %1" :: "s"(g0), "s"(g1) : "memory");
    __builtin_amdgcn_s_wait_tensorcnt(0);
  }
  asm volatile("s_wait_asynccnt 0" ::: "memory");
  __syncthreads();

  for (int k = tid; k < EMBD; k += 256) hsf[k] += b1[k];
  if (tid < 64) score[tid] = b3[0];
  __syncthreads();

  // ---- build A = bf16(gelu(h_doc[d,k] + hsf[k])) in LDS ----
  const float* hb = h + (size_t)b * 96 * EMBD;
  for (int it = 0; it < 48; ++it) {
    int e = (it * 256 + tid) * 4;
    int d = e / EMBD, k = e % EMBD;
    float4 x = *(const float4*)(hb + (size_t)d * EMBD + k);
    uint2 p;
    p.x = pack_bf2(gelu_exact(x.x + hsf[k]),     gelu_exact(x.y + hsf[k + 1]));
    p.y = pack_bf2(gelu_exact(x.z + hsf[k + 2]), gelu_exact(x.w + hsf[k + 3]));
    *(uint2*)(A + (size_t)d * A_STRIDE + k) = p;
  }
  __syncthreads();

  // ---- GEMM h2 = gelu(A[64x768] @ W2[768x384] + b2); fold W3 dot into epilogue ----
  const int wave = tid >> 5, lane = tid & 31;
  {
    const int half = lane >> 4, m = lane & 15;
    for (int c = 0; c < 3; ++c) {
      const int ct = wave * 3 + c;                // 8 waves x 3 = 24 coltiles
      const uint4* brow = (const uint4*)(w2t + (size_t)(ct * 16 + m) * EMBD);
      v8f acc[4];
#pragma unroll
      for (int rt = 0; rt < 4; ++rt) acc[rt] = (v8f){0.f,0.f,0.f,0.f,0.f,0.f,0.f,0.f};
#pragma unroll 4
      for (int k = 0; k < 24; ++k) {
        const int kk = k * 32;
        Frag bu;
        const int bb = kk + half * 16;
        bu.q[0] = brow[bb >> 3];
        bu.q[1] = brow[(bb >> 3) + 1];
#pragma unroll
        for (int rt = 0; rt < 4; ++rt) {          // reuse B fragment across 4 rowtiles
          Frag au;
          const unsigned short* arow = A + (size_t)(rt * 16 + m) * A_STRIDE;
          au.q[0] = *(const uint4*)(arow + kk + half * 16);
          au.q[1] = *(const uint4*)(arow + kk + half * 16 + 8);
          acc[rt] = __builtin_amdgcn_wmma_f32_16x16x32_bf16(false, au.v, false, bu.v, (short)0, acc[rt], false, false);
        }
      }
      const int N = ct * 16 + m;
      const float w3n = W3[N];
      const float b2n = b2[N];
#pragma unroll
      for (int rt = 0; rt < 4; ++rt) {
#pragma unroll
        for (int r = 0; r < 8; ++r) {
          const int M = rt * 16 + r + 8 * half;
          float c2 = gelu_exact(acc[rt][r] + b2n) * w3n;
          c2 += __shfl_xor(c2, 1, 32);
          c2 += __shfl_xor(c2, 2, 32);
          c2 += __shfl_xor(c2, 4, 32);
          c2 += __shfl_xor(c2, 8, 32);             // reduce 16 lanes sharing M
          if (m == 0) atomicAdd(&score[M], c2);    // ds_add_f32
        }
      }
    }
  }
  __syncthreads();

  // ---- sigmoid + doc mask + max over d -> pred ----
  const int nd = ndocp[b];
  if (tid < 64) score[tid] = (tid < nd) ? (1.f / (1.f + expf(-score[tid]))) : 0.f;
  __syncthreads();
  if (tid < 32) {
    float mx = fmaxf(score[tid], score[tid + 32]);
    mx = fmaxf(mx, __shfl_xor(mx, 1, 32));
    mx = fmaxf(mx, __shfl_xor(mx, 2, 32));
    mx = fmaxf(mx, __shfl_xor(mx, 4, 32));
    mx = fmaxf(mx, __shfl_xor(mx, 8, 32));
    mx = fmaxf(mx, __shfl_xor(mx, 16, 32));
    if (tid == 0) {
      float segw = (nsegp[b] > 10) ? 1.0f : 100.0f;
      float p = segw * mx;
      pv[0] = p;
      out[1056 + b * 32 + s] = p;                  // pred slot
    }
  }
  __syncthreads();

  // ---- cosine prior: dscos = max_d mask*(1 - cos(doc_d, sum_s))/2 ----
  float lmax = 0.f;
  for (int i = 0; i < 8; ++i) {
    int d = wave + 8 * i;
    const float* dr = doc + ((size_t)b * 64 + d) * EMBD;
    float dot = 0.f, dnn = 0.f, snn = 0.f;
    for (int k = lane; k < EMBD; k += 32) {
      float dv = dr[k], sv = ssf[k];
      dot = fmaf(dv, sv, dot);
      dnn = fmaf(dv, dv, dnn);
      snn = fmaf(sv, sv, snn);
    }
#pragma unroll
    for (int msk = 1; msk < 32; msk <<= 1) {
      dot += __shfl_xor(dot, msk, 32);
      dnn += __shfl_xor(dnn, msk, 32);
      snn += __shfl_xor(snn, msk, 32);
    }
    if (d < nd) {
      float inv = rsqrtf(fmaxf(dnn, 1e-12f)) * rsqrtf(fmaxf(snn, 1e-12f));
      lmax = fmaxf(lmax, 0.5f * (1.f - dot * inv));
    }
  }
  if (lane == 0) wmaxa[wave] = lmax;
  __syncthreads();
  if (tid == 0) {
    float dmx = 0.f;
#pragma unroll
    for (int i = 0; i < 8; ++i) dmx = fmaxf(dmx, wmaxa[i]);
    diffws[b * 32 + s] = (s < nsegp[b]) ? fabsf(dmx - pv[0]) : 0.f;
  }
}

// =====================================================================
// Kernel 4: remaining outputs. Layout: [0,32) agg | [32,1056) target_inst |
// [1056,2080) pred | [2080,2112) num_seg (int bits) | [2112] cos_prior
// =====================================================================
__global__ __launch_bounds__(1024) void final_kernel(
    const float* __restrict__ tagg, const float* __restrict__ tinst,
    const int* __restrict__ nseg, const float* __restrict__ diffws,
    float* __restrict__ out) {
  __shared__ float accs[32];
  const int tid = threadIdx.x;
  if (tid < 1024) {
    int b = tid >> 5, s = tid & 31;
    out[32 + tid] = (s < nseg[b]) ? tinst[tid] : 1.0f;
  }
  if (tid < 32) {
    out[tid] = tagg[tid];
    ((int*)out)[2080 + tid] = nseg[tid];
    float ssum = 0.f;
    for (int s = 0; s < 32; ++s) ssum += diffws[tid * 32 + s];
    accs[tid] = ssum / (float)nseg[tid];
  }
  __syncthreads();
  if (tid == 0) {
    float t = 0.f;
    for (int i = 0; i < 32; ++i) t += accs[i];
    out[2112] = t * (1.0f / 32.0f);
  }
}

// =====================================================================
extern "C" void kernel_launch(void* const* d_in, const int* in_sizes, int n_in,
                              void* d_out, int out_size, void* d_ws, size_t ws_size,
                              hipStream_t stream) {
  const float* doc   = (const float*)d_in[0];
  const float* sum   = (const float*)d_in[1];
  const int*   ndoc  = (const int*)d_in[2];
  const int*   nseg  = (const int*)d_in[3];
  const float* tagg  = (const float*)d_in[4];
  const float* tinst = (const float*)d_in[5];
  const float* W1    = (const float*)d_in[6];
  const float* b1    = (const float*)d_in[7];
  const float* W2    = (const float*)d_in[8];
  const float* b2    = (const float*)d_in[9];
  const float* W3    = (const float*)d_in[10];
  const float* b3    = (const float*)d_in[11];

  // workspace layout (~12.4 MB total)
  char* ws = (char*)d_ws;
  unsigned short* w1t = (unsigned short*)ws;                         // 768*1536 bf16
  unsigned short* w2t = w1t + (size_t)768 * 1536;                    // 384*768 bf16
  float* h      = (float*)(ws + ((size_t)768 * 1536 + (size_t)384 * 768) * 2); // 32*96*768 f32
  float* diffws = h + (size_t)32 * 96 * 768;                         // 1024 f32

  const int n_conv = 768 * 1536 + 384 * 768;
  convert_w_kernel<<<(n_conv + 255) / 256, 256, 0, stream>>>(W1, W2, w1t, w2t);
  h_gemm_kernel<<<dim3(32, 6), 256, 0, stream>>>(doc, sum, w1t, h);
  fused_kernel<<<dim3(32, 32), 256, FUSED_LDS, stream>>>(h, doc, sum, w2t, ndoc, nseg,
                                                         b1, b2, W3, b3, (float*)d_out, diffws);
  final_kernel<<<1, 1024, 0, stream>>>(tagg, tinst, nseg, diffws, (float*)d_out);
}
